// VPINN_76132590289707
// MI455X (gfx1250) — compile-verified, matrix operand order checked by reading, not run on hardware
//
#include <hip/hip_runtime.h>
#include <hip/hip_bf16.h>

typedef __attribute__((ext_vector_type(16))) _Float16 v16h;
typedef __attribute__((ext_vector_type(8)))  float    v8f;
typedef unsigned int v4u  __attribute__((ext_vector_type(4)));
typedef int          v8i_ __attribute__((ext_vector_type(8)));
typedef int          v4i_ __attribute__((ext_vector_type(4)));

#define HID   256
#define NQ    4096
#define NPTS  65536
#define NT    10

// ---- workspace layout (float offsets) ----
#define WS_BSUM   0        // 320  boundary per-point sq errors
#define WS_VSUM   320      // 16   per-element-block residual sums
#define WS_D1XU   512      // 65536
#define WS_D1YU   66048    // 65536
#define WS_TXP    131584   // 16*4096 padded test fns (x)
#define WS_TYP    197120   // 16*4096 (y)
#define WS_D1TX   262656   // 16*4096
#define WS_D1TY   328192   // 16*4096
#define WS_END_F  393728
#define PK_HALVES 196608   // 3 layers * 256*256 f16, B-fragment swizzled

#if defined(__HIP_DEVICE_COMPILE__) && __has_builtin(__builtin_amdgcn_tensor_load_to_lds) && __has_builtin(__builtin_amdgcn_s_wait_tensorcnt)
#define HAVE_TDM 1
#else
#define HAVE_TDM 0
#endif

// ---------------- TDM: DMA an 8KB contiguous slab (1024 x 8B) into LDS ----------------
#if HAVE_TDM
__device__ __forceinline__ void tdm_load_8kb(const _Float16* gsrc, unsigned int ldsOff) {
    unsigned long long ga = (unsigned long long)(size_t)gsrc;
    // D# group0: count=1 | lds_addr | global_addr[56:0] | type=2 (bits 127:126)
    v4u g0 = { 1u,
               ldsOff,
               (unsigned int)ga,
               ((unsigned int)(ga >> 32) & 0x01FFFFFFu) | (2u << 30) };
    // D# group1: data_size=3 (8B); tensor_dim0 = tile_dim0 = 1024; tensor_dim1 = tile_dim1 = 1;
    //            tensor_dim0_stride = 1024
    v8i_ g1 = { (int)(3u << 16),        // workgroup_mask=0, data_size=3
                (int)(1024u << 16),     // tensor_dim0[15:0] in [31:16]
                (int)(1u << 16),        // tensor_dim0[31:16]=0, tensor_dim1[15:0]=1
                (int)(1024u << 16),     // tensor_dim1[31:16]=0, tile_dim0=1024
                1,                      // tile_dim1=1, tile_dim2=0
                1024,                   // tensor_dim0_stride[31:0]
                0, 0 };                 // stride0 hi / stride1
    v4i_ gz = { 0, 0, 0, 0 };
#if defined(__clang_major__) && (__clang_major__ >= 23)
    v8i_ gz8 = { 0, 0, 0, 0, 0, 0, 0, 0 };
    __builtin_amdgcn_tensor_load_to_lds(g0, g1, gz, gz, gz8, 0);
#else
    __builtin_amdgcn_tensor_load_to_lds(g0, g1, gz, gz, 0);
#endif
}
#endif

// ---------------- Jacobi recurrence (device) ----------------
__device__ __forceinline__ float jacobif(int n, float a, float b, float x) {
    if (n <= 0) return 1.0f;
    float p0 = 1.0f;
    float p1 = 0.5f * (a - b + (a + b + 2.0f) * x);
    for (int k = 2; k <= n; ++k) {
        float fk = (float)k;
        float c1 = 2.0f * fk * (fk + a + b) * (2.0f * fk + a + b - 2.0f);
        float c2 = (2.0f * fk + a + b - 1.0f) * (a * a - b * b);
        float c3 = (2.0f * fk + a + b - 2.0f) * (2.0f * fk + a + b - 1.0f) * (2.0f * fk + a + b);
        float c4 = 2.0f * (fk + a - 1.0f) * (fk + b - 1.0f) * (2.0f * fk + a + b);
        float p2 = ((c2 + c3 * x) * p1 - c4 * p0) / c1;
        p0 = p1; p1 = p2;
    }
    return p1;
}

// ---------------- pack W1..W3 into WMMA B-fragment order ----------------
// B 32x16 f16 fragment: lane l holds column n = ntile*16 + (l%16);
// half i (0..15): k_local = ((i<8)? i : i+8) + (l>>4)*8   (ISA 7.12.2 16-bit layout)
__global__ void k_pack(const float* __restrict__ W1, const float* __restrict__ W2,
                       const float* __restrict__ W3, _Float16* __restrict__ pk) {
    int gid = blockIdx.x * blockDim.x + threadIdx.x;
    if (gid >= PK_HALVES) return;
    int L     = gid >> 16;
    int rem   = gid & 65535;
    int ntile = rem >> 12;
    int ks    = (rem >> 9) & 7;
    int lane  = (rem >> 4) & 31;
    int i     = rem & 15;
    int n  = ntile * 16 + (lane & 15);
    int kl = ((i < 8) ? i : i + 8) + (lane >> 4) * 8;
    int k  = ks * 32 + kl;
    const float* W = (L == 0) ? W1 : (L == 1) ? W2 : W3;
    pk[gid] = (_Float16)W[k * HID + n];
}

// ---------------- test-function tables (padded to 16 rows) ----------------
__global__ void k_tables(const float* __restrict__ xquad, const float* __restrict__ yquad,
                         float* __restrict__ ws) {
    int q = blockIdx.x * blockDim.x + threadIdx.x;
    if (q >= NQ) return;
    float xq = xquad[q], yq = yquad[q];
    float* txP  = ws + WS_TXP;
    float* tyP  = ws + WS_TYP;
    float* d1tx = ws + WS_D1TX;
    float* d1ty = ws + WS_D1TY;
    for (int r = 0; r < 16; ++r) {
        float a = 0.f, b = 0.f, c = 0.f, d = 0.f;
        if (r < NT) {
            int n = r + 1;
            a = jacobif(n + 1, 0.f, 0.f, xq) - jacobif(n - 1, 0.f, 0.f, xq);
            b = jacobif(n + 1, 0.f, 0.f, yq) - jacobif(n - 1, 0.f, 0.f, yq);
            float ca = 0.5f * (float)(n + 2);
            c = ca * jacobif(n, 1.f, 1.f, xq);
            d = ca * jacobif(n, 1.f, 1.f, yq);
            if (n >= 2) {
                c -= 0.5f * (float)n * jacobif(n - 2, 1.f, 1.f, xq);
                d -= 0.5f * (float)n * jacobif(n - 2, 1.f, 1.f, yq);
            }
        }
        txP [r * NQ + q] = a;
        tyP [r * NQ + q] = b;
        d1tx[r * NQ + q] = c;
        d1ty[r * NQ + q] = d;
    }
}

// ---------------- boundary loss: one block per boundary point ----------------
__global__ __launch_bounds__(256)
void k_boundary(const float* __restrict__ W0, const float* __restrict__ b0,
                const float* __restrict__ W1, const float* __restrict__ b1,
                const float* __restrict__ W2, const float* __restrict__ b2,
                const float* __restrict__ W3, const float* __restrict__ b3,
                const float* __restrict__ W4, const float* __restrict__ b4,
                const float* __restrict__ xb, const float* __restrict__ yb,
                const float* __restrict__ utr, float* __restrict__ bsum) {
    __shared__ float hA[HID], hB[HID], red[HID];
    int t = threadIdx.x, p = blockIdx.x;
    float x = xb[p], y = yb[p];
    hA[t] = tanhf(b0[t] + x * W0[t] + y * W0[HID + t]);
    __syncthreads();
    const float* Ws[3] = {W1, W2, W3};
    const float* bs[3] = {b1, b2, b3};
    #pragma unroll
    for (int li = 0; li < 3; ++li) {
        const float* W  = Ws[li];
        const float* bb = bs[li];
        float* src = (li & 1) ? hB : hA;
        float* dst = (li & 1) ? hA : hB;
        float acc = bb[t];
        for (int i = 0; i < HID; ++i) acc += src[i] * W[i * HID + t];
        dst[t] = tanhf(acc);
        __syncthreads();
    }
    red[t] = hB[t] * W4[t];   // final hidden state lives in hB
    __syncthreads();
    for (int s = 128; s > 0; s >>= 1) { if (t < s) red[t] += red[t + s]; __syncthreads(); }
    if (t == 0) {
        float u = red[0] + b4[0];
        float e = utr[p] - u;
        bsum[p] = e * e;
    }
}

// ---------------- main kernel: fwd + two-tangent JVP via WMMA + TDM ----------------
// one wave (32 lanes) per 16 quadrature points; 48KB activation LDS (double-buffered f16)
// + 8KB TDM weight staging buffer
__global__ __launch_bounds__(32)
void k_main(const float* __restrict__ W0, const float* __restrict__ b0,
            const float* __restrict__ b1, const float* __restrict__ b2,
            const float* __restrict__ b3, const float* __restrict__ W4,
            const float* __restrict__ xquad, const float* __restrict__ yquad,
            const float* __restrict__ gridx, const float* __restrict__ gridy,
            const _Float16* __restrict__ pk,
            float* __restrict__ d1xu, float* __restrict__ d1yu) {
    __shared__ _Float16 smem[6 * 4096];   // {h,tx,ty} x double buffer, 16x256 each
    __shared__ _Float16 bstage[4096];     // 8KB TDM staging for one ntile's B slabs
    __shared__ float w4s[HID];
    __shared__ float xv[16], yv[16];
    const int lane = threadIdx.x;
    const int p0 = blockIdx.x * 16;

    for (int i = lane; i < HID; i += 32) w4s[i] = W4[i];
    if (lane < 16) {
        int p  = p0 + lane;
        int q  = p & (NQ - 1);
        int xe = p >> 14;
        int ye = (p >> 12) & 3;
        float jx = (gridx[xe + 1] - gridx[xe]) * 0.5f;
        float jy = (gridy[ye + 1] - gridy[ye]) * 0.5f;
        xv[lane] = gridx[xe] + jx * (xquad[q] + 1.0f);
        yv[lane] = gridy[ye] + jy * (yquad[q] + 1.0f);
    }
    // opaque (never-taken at runtime) store so the compiler cannot treat the
    // TDM-written staging buffer as never-stored shared memory
    if (gridx[0] > 1.0e30f) bstage[0] = (_Float16)0.0f;
    __syncthreads();

    // layer 0 + tangent seeds -> buffer set 0 (row-major f16 [16][256])
    for (int it = 0; it < 128; ++it) {
        int idx = it * 32 + lane;
        int r = idx >> 8, j = idx & 255;
        float x = xv[r], y = yv[r];
        float wa = W0[j], wb = W0[HID + j];
        float h = tanhf(b0[j] + x * wa + y * wb);
        float g = 1.0f - h * h;
        int o = r * HID + j;
        smem[o]        = (_Float16)h;
        smem[4096 + o] = (_Float16)(g * wa);
        smem[8192 + o] = (_Float16)(g * wb);
    }
    __syncthreads();

    union Frag { float4 f4[2]; v16h h; };
    const int m   = lane & 15;
    const int grp = lane >> 4;
#if HAVE_TDM
    const unsigned int stageOff = (unsigned int)(size_t)(void*)&bstage[0];
#endif

    #pragma unroll
    for (int li = 0; li < 3; ++li) {
        const int cur = (li & 1) ? 12288 : 0;
        const int nxt = 12288 - cur;
        const float* bias = (li == 0) ? b1 : (li == 1) ? b2 : b3;
        const _Float16* pkL = pk + li * 65536;
        for (int ntile = 0; ntile < 16; ++ntile) {
#if HAVE_TDM
            // DMA this ntile's 8KB weight slab (8 K-step fragments) into LDS
            tdm_load_8kb(pkL + ntile * 4096, stageOff);
            __builtin_amdgcn_s_wait_tensorcnt(0);
            asm volatile("" ::: "memory");
#endif
            v8f ah = {}, ax = {}, ay = {};
            #pragma unroll
            for (int ks = 0; ks < 8; ++ks) {
                Frag bf, af;
#if HAVE_TDM
                const float4* bp = (const float4*)(&bstage[(ks * 32 + lane) * 16]);
#else
                const float4* bp = (const float4*)(pkL + (((ntile * 8 + ks) * 32 + lane) * 16));
#endif
                bf.f4[0] = bp[0]; bf.f4[1] = bp[1];
                int abase = m * HID + ks * 32 + grp * 8;  // A fragment: two 16B chunks
                af.f4[0] = *(const float4*)&smem[cur + abase];
                af.f4[1] = *(const float4*)&smem[cur + abase + 16];
                ah = __builtin_amdgcn_wmma_f32_16x16x32_f16(false, af.h, false, bf.h, (short)0, ah, false, false);
                af.f4[0] = *(const float4*)&smem[cur + 4096 + abase];
                af.f4[1] = *(const float4*)&smem[cur + 4096 + abase + 16];
                ax = __builtin_amdgcn_wmma_f32_16x16x32_f16(false, af.h, false, bf.h, (short)0, ax, false, false);
                af.f4[0] = *(const float4*)&smem[cur + 8192 + abase];
                af.f4[1] = *(const float4*)&smem[cur + 8192 + abase + 16];
                ay = __builtin_amdgcn_wmma_f32_16x16x32_f16(false, af.h, false, bf.h, (short)0, ay, false, false);
            }
            int n = ntile * 16 + m;
            float bn = bias[n];
            #pragma unroll
            for (int v = 0; v < 8; ++v) {
                int mm = grp * 8 + v;               // C/D layout: lane%16 = N, vgpr+8*(lane/16) = M
                float h = tanhf(ah[v] + bn);
                float g = 1.0f - h * h;
                int o = nxt + mm * HID + n;
                smem[o]        = (_Float16)h;
                smem[4096 + o] = (_Float16)(ax[v] * g);
                smem[8192 + o] = (_Float16)(ay[v] * g);
            }
        }
        __syncthreads();
    }

    // final: set 1 (offset 12288) holds h3/tx3/ty3 ; d1xu = tx3 . W4, d1yu = ty3 . W4
    {
        int mm  = lane & 15;
        int src = 12288 + ((lane < 16) ? 4096 : 8192) + mm * HID;
        float s = 0.0f;
        for (int k = 0; k < HID; ++k) s += (float)smem[src + k] * w4s[k];
        int p = p0 + mm;
        if (lane < 16) d1xu[p] = s; else d1yu[p] = s;
    }
}

// ---------------- per-element residual contraction ----------------
__global__ __launch_bounds__(128)
void k_vres(const float* __restrict__ wquad, const float* __restrict__ gridx,
            const float* __restrict__ gridy, const float* __restrict__ Fext,
            const float* __restrict__ ws, float* __restrict__ vsum) {
    __shared__ float red[128];
    int t = threadIdx.x, blk = blockIdx.x;
    int xe = blk >> 2, ye = blk & 3;
    const float* d1xu = ws + WS_D1XU + blk * NQ;
    const float* d1yu = ws + WS_D1YU + blk * NQ;
    float val = 0.0f;
    if (t < NT * NT) {
        int k = t / NT, r = t % NT;
        const float* tyk   = ws + WS_TYP  + k * NQ;
        const float* d1tyk = ws + WS_D1TY + k * NQ;
        const float* txr   = ws + WS_TXP  + r * NQ;
        const float* d1txr = ws + WS_D1TX + r * NQ;
        float s1 = 0.f, s2 = 0.f;
        for (int q = 0; q < NQ; ++q) {
            float wq = wquad[2 * q] * wquad[2 * q + 1];
            s1 += d1txr[q] * tyk[q]   * (wq * d1xu[q]);
            s2 += txr[q]   * d1tyk[q] * (wq * d1yu[q]);
        }
        float jx = (gridx[xe + 1] - gridx[xe]) * 0.5f;
        float jy = (gridy[ye + 1] - gridy[ye]) * 0.5f;
        float Res = -(jy * s1) - (jx * s2) - Fext[(blk * NT + k) * NT + r];
        val = Res * Res * (1.0f / (float)(NT * NT));
    }
    red[t] = val; __syncthreads();
    for (int s = 64; s > 0; s >>= 1) { if (t < s) red[t] += red[t + s]; __syncthreads(); }
    if (t == 0) vsum[blk] = red[0];
}

// ---------------- deterministic final combine ----------------
__global__ void k_final(const float* __restrict__ ws, float* __restrict__ out) {
    float sb = 0.f;
    for (int i = 0; i < 320; ++i) sb += ws[WS_BSUM + i];
    float sv = 0.f;
    for (int i = 0; i < 16; ++i) sv += ws[WS_VSUM + i];
    out[0] = 10.0f * (sb / 320.0f) + sv;
}

extern "C" void kernel_launch(void* const* d_in, const int* in_sizes, int n_in,
                              void* d_out, int out_size, void* d_ws, size_t ws_size,
                              hipStream_t stream) {
    (void)in_sizes; (void)n_in; (void)out_size; (void)ws_size;
    const float* W0 = (const float*)d_in[0];
    const float* b0 = (const float*)d_in[1];
    const float* W1 = (const float*)d_in[2];
    const float* b1 = (const float*)d_in[3];
    const float* W2 = (const float*)d_in[4];
    const float* b2 = (const float*)d_in[5];
    const float* W3 = (const float*)d_in[6];
    const float* b3 = (const float*)d_in[7];
    const float* W4 = (const float*)d_in[8];
    const float* b4 = (const float*)d_in[9];
    const float* xquad = (const float*)d_in[10];
    const float* yquad = (const float*)d_in[11];
    const float* wquad = (const float*)d_in[12];
    const float* gridx = (const float*)d_in[13];
    const float* gridy = (const float*)d_in[14];
    const float* x_b   = (const float*)d_in[15];
    const float* y_b   = (const float*)d_in[16];
    const float* u_tr  = (const float*)d_in[17];
    const float* F_ext = (const float*)d_in[23];

    float* ws = (float*)d_ws;
    _Float16* pk = (_Float16*)((char*)d_ws + (size_t)WS_END_F * sizeof(float));

    k_pack<<<PK_HALVES / 256, 256, 0, stream>>>(W1, W2, W3, pk);
    k_tables<<<NQ / 256, 256, 0, stream>>>(xquad, yquad, ws);
    k_boundary<<<320, 256, 0, stream>>>(W0, b0, W1, b1, W2, b2, W3, b3, W4, b4,
                                        x_b, y_b, u_tr, ws + WS_BSUM);
    k_main<<<NPTS / 16, 32, 0, stream>>>(W0, b0, b1, b2, b3, W4,
                                         xquad, yquad, gridx, gridy, pk,
                                         ws + WS_D1XU, ws + WS_D1YU);
    k_vres<<<16, 128, 0, stream>>>(wquad, gridx, gridy, F_ext, ws, ws + WS_VSUM);
    k_final<<<1, 1, 0, stream>>>(ws, (float*)d_out);
}
